// MultiHead_39651138076763
// MI455X (gfx1250) — compile-verified, matrix operand order checked by reading, not run on hardware
//
#include <hip/hip_runtime.h>
#include <hip/hip_bf16.h>

#define B_DIM 4
#define T_DIM 2048
#define E_DIM 1024
#define H_DIM 16
#define D_DIM 64
#define HD_DIM 1024
#define QK_SCALE 0.125f   // 1/sqrt(64)

typedef __attribute__((ext_vector_type(16))) _Float16 v16h;
typedef __attribute__((ext_vector_type(8)))  _Float16 v8h;
typedef __attribute__((ext_vector_type(8)))  float    v8f;

__device__ __forceinline__ v8f vzero8f() {
  v8f z = {0.f, 0.f, 0.f, 0.f, 0.f, 0.f, 0.f, 0.f};
  return z;
}

// Load a 16x32 f16 A-style fragment (also valid as B-fragment for row-major
// "transposed-B" operands). Row-major source, stride `ld` halves.
// Wave32 layout (ISA 7.12.2): lane holds row (lane&15); K indices:
// frag[i] = src[row][ (i<8 ? kb+i : 16+kb+(i-8)) ],  kb = (lane>>4)*8.
// => two contiguous 16B loads per lane (global_load_b128 / ds_load_b128).
__device__ __forceinline__ v16h load_frag(const _Float16* base, int ld, int lane) {
  const int r  = lane & 15;
  const int kb = (lane >> 4) << 3;
  const _Float16* p = base + (size_t)r * (size_t)ld + kb;
  v8h lo = *(const v8h*)(p);
  v8h hi = *(const v8h*)(p + 16);
  return __builtin_shufflevector(lo, hi, 0, 1, 2, 3, 4, 5, 6, 7,
                                 8, 9, 10, 11, 12, 13, 14, 15);
}

__device__ __forceinline__ v8f wmma_f16f32(v16h a, v16h b, v8f c) {
  // (neg_a, A, neg_b, B, c_mod, C, reuse_a, reuse_b)
  return __builtin_amdgcn_wmma_f32_16x16x32_f16(false, a, false, b, (short)0, c,
                                                false, false);
}

// ---------------- prep kernels ----------------

__global__ void __launch_bounds__(256) cvt_f32_to_f16(const float* __restrict__ src,
                                                      _Float16* __restrict__ dst,
                                                      int n) {
  int i = blockIdx.x * 256 + threadIdx.x;
  if (i < n) dst[i] = (_Float16)src[i];
}

// W [H,E,D] f32 -> Wt [H,D,E] f16
__global__ void __launch_bounds__(256) cvt_tr_hed(const float* __restrict__ W,
                                                  _Float16* __restrict__ Wt) {
  int i = blockIdx.x * 256 + threadIdx.x;  // over H*D*E
  if (i >= H_DIM * D_DIM * E_DIM) return;
  int e = i % E_DIM;
  int d = (i / E_DIM) % D_DIM;
  int h = i / (E_DIM * D_DIM);
  Wt[i] = (_Float16)W[((size_t)h * E_DIM + e) * D_DIM + d];
}

// Wo [HD,E] f32 -> Wot [E,HD] f16
__global__ void __launch_bounds__(256) cvt_tr_wo(const float* __restrict__ W,
                                                 _Float16* __restrict__ Wt) {
  int i = blockIdx.x * 256 + threadIdx.x;  // over E*HD
  if (i >= E_DIM * HD_DIM) return;
  int c = i % HD_DIM;   // hd (contraction) index
  int e = i / HD_DIM;   // output-embedding row of Wot
  Wt[i] = (_Float16)W[(size_t)c * E_DIM + e];
}

// ---------------- QKV projection ----------------
// One wave -> 16 rows x 64 cols of q, k, v for one (b,h).
__global__ void __launch_bounds__(256) qkv_kernel(
    const _Float16* __restrict__ xh,    // [B*T, E]
    const _Float16* __restrict__ Wqt,   // [H, D, E]
    const _Float16* __restrict__ Wkt,   // [H, D, E]
    const _Float16* __restrict__ Wvt,   // [H, D, E]
    _Float16* __restrict__ qd,          // [B, H, T, D]  (pre-scaled by 1/sqrt(D))
    _Float16* __restrict__ kd,          // [B, H, T, D]
    _Float16* __restrict__ vtd) {       // [B, H, D, T]  (transposed for PV gemm)
  const int lane = threadIdx.x & 31;
  const int wave = blockIdx.x * 8 + (threadIdx.x >> 5);
  const int tilesPerBH = T_DIM / 16;  // 128
  const int tile = wave % tilesPerBH;
  const int bh = wave / tilesPerBH;
  const int h = bh % H_DIM;
  const int b = bh / H_DIM;
  if (b >= B_DIM) return;
  const int t0 = tile * 16;

  const _Float16* xrow = xh + ((size_t)b * T_DIM + t0) * E_DIM;
  const _Float16* wq = Wqt + (size_t)h * D_DIM * E_DIM;
  const _Float16* wk = Wkt + (size_t)h * D_DIM * E_DIM;
  const _Float16* wv = Wvt + (size_t)h * D_DIM * E_DIM;

  v8f aq[4], ak[4], av[4];
#pragma unroll
  for (int nt = 0; nt < 4; ++nt) {
    aq[nt] = vzero8f(); ak[nt] = vzero8f(); av[nt] = vzero8f();
  }

  for (int kk = 0; kk < E_DIM; kk += 32) {
    v16h afrag = load_frag(xrow + kk, E_DIM, lane);  // reused by 12 WMMAs
#pragma unroll
    for (int nt = 0; nt < 4; ++nt) {
      v16h bq = load_frag(wq + (size_t)(nt * 16) * E_DIM + kk, E_DIM, lane);
      aq[nt] = wmma_f16f32(afrag, bq, aq[nt]);
      v16h bk = load_frag(wk + (size_t)(nt * 16) * E_DIM + kk, E_DIM, lane);
      ak[nt] = wmma_f16f32(afrag, bk, ak[nt]);
      v16h bv = load_frag(wv + (size_t)(nt * 16) * E_DIM + kk, E_DIM, lane);
      av[nt] = wmma_f16f32(afrag, bv, av[nt]);
    }
  }

  const int col = lane & 15;
  const int rbase = (lane >> 4) * 8;  // C-frag: M = r + 8*(lane>>4)
  const size_t qkbase = (((size_t)b * H_DIM + h) * T_DIM + t0) * D_DIM;
  const size_t vtbase = ((size_t)b * H_DIM + h) * (size_t)D_DIM * T_DIM;
#pragma unroll
  for (int nt = 0; nt < 4; ++nt) {
#pragma unroll
    for (int r = 0; r < 8; ++r) {
      const int row = rbase + r;
      const int d = nt * 16 + col;
      qd[qkbase + (size_t)row * D_DIM + d] = (_Float16)(aq[nt][r] * QK_SCALE);
      kd[qkbase + (size_t)row * D_DIM + d] = (_Float16)(ak[nt][r]);
      vtd[vtbase + (size_t)d * T_DIM + (t0 + row)] = (_Float16)(av[nt][r]);
    }
  }
}

// ---------------- flash attention (causal, online softmax) ----------------
// One wave -> one 16-row Q tile for one (b,h); iterates S in 32-col chunks.
__global__ void __launch_bounds__(256) flash_attn_kernel(
    const _Float16* __restrict__ qd,    // [B,H,T,D]
    const _Float16* __restrict__ kd,    // [B,H,T,D]
    const _Float16* __restrict__ vtd,   // [B,H,D,T]
    _Float16* __restrict__ attn) {      // [B,T,H*D]
  __shared__ _Float16 plds[8][16][32];  // per-wave P transpose staging (8 KB)
  const int lane = threadIdx.x & 31;
  const int wslot = threadIdx.x >> 5;
  const int wave = blockIdx.x * 8 + wslot;
  const int tilesPerBH = T_DIM / 16;
  const int tile = wave % tilesPerBH;
  const int bh = wave / tilesPerBH;
  const int h = bh % H_DIM;
  const int b = bh / H_DIM;
  if (b >= B_DIM) return;
  const int t0 = tile * 16;

  const _Float16* qp = qd + (((size_t)b * H_DIM + h) * T_DIM + t0) * D_DIM;
  const _Float16* kp = kd + ((size_t)b * H_DIM + h) * (size_t)T_DIM * D_DIM;
  const _Float16* vp = vtd + ((size_t)b * H_DIM + h) * (size_t)D_DIM * T_DIM;

  // Q fragments for K-dim 64 = two K=32 fragments; live for the whole loop.
  const v16h aq0 = load_frag(qp, D_DIM, lane);
  const v16h aq1 = load_frag(qp + 32, D_DIM, lane);

  v8f o[4];
#pragma unroll
  for (int nt = 0; nt < 4; ++nt) o[nt] = vzero8f();
  float mrow[8], lrow[8];
#pragma unroll
  for (int r = 0; r < 8; ++r) { mrow[r] = -1e30f; lrow[r] = 0.f; }

  const int col = lane & 15;
  const int rbase = (lane >> 4) * 8;
  const int smax = t0 + 15;  // largest valid column for this row tile

  for (int s0 = 0; s0 <= smax; s0 += 32) {
    // ---- scores: two 16-col tiles, K=64 contraction (2 WMMAs each) ----
    v8f sc[2];
#pragma unroll
    for (int ct = 0; ct < 2; ++ct) {
      sc[ct] = vzero8f();
      const _Float16* kb = kp + (size_t)(s0 + ct * 16) * D_DIM;  // Kt cols = K rows
      v16h b0 = load_frag(kb, D_DIM, lane);
      sc[ct] = wmma_f16f32(aq0, b0, sc[ct]);
      v16h b1 = load_frag(kb + 32, D_DIM, lane);
      sc[ct] = wmma_f16f32(aq1, b1, sc[ct]);
    }
    // ---- causal mask (per-lane select; no EXEC change) ----
#pragma unroll
    for (int ct = 0; ct < 2; ++ct) {
      const int sg = s0 + ct * 16 + col;
#pragma unroll
      for (int r = 0; r < 8; ++r) {
        const int trow = t0 + rbase + r;
        float v = sc[ct][r];
        sc[ct][r] = (sg > trow) ? -1e30f : v;
      }
    }
    // ---- online softmax update (reductions within 16-lane halves) ----
    float alpha[8];
#pragma unroll
    for (int r = 0; r < 8; ++r) {
      float mx = fmaxf(sc[0][r], sc[1][r]);
#pragma unroll
      for (int off = 8; off >= 1; off >>= 1)
        mx = fmaxf(mx, __shfl_xor(mx, off, 32));
      const float mn = fmaxf(mrow[r], mx);
      alpha[r] = __expf(mrow[r] - mn);
      mrow[r] = mn;
      float p0 = __expf(sc[0][r] - mn);
      float p1 = __expf(sc[1][r] - mn);
      sc[0][r] = p0;
      sc[1][r] = p1;
      float ps = p0 + p1;
#pragma unroll
      for (int off = 8; off >= 1; off >>= 1) ps += __shfl_xor(ps, off, 32);
      lrow[r] = lrow[r] * alpha[r] + ps;
#pragma unroll
      for (int nt = 0; nt < 4; ++nt) o[nt][r] = o[nt][r] * alpha[r];
    }
    // ---- transpose P through LDS: C-layout -> A-fragment layout ----
#pragma unroll
    for (int ct = 0; ct < 2; ++ct)
#pragma unroll
      for (int r = 0; r < 8; ++r)
        plds[wslot][rbase + r][ct * 16 + col] = (_Float16)sc[ct][r];
    __asm__ volatile("s_wait_dscnt 0" ::: "memory");
    const v16h pa = load_frag(&plds[wslot][0][0], 32, lane);
    // ---- O += P @ V  (V pre-transposed: B-frag = A-style load of Vt) ----
#pragma unroll
    for (int nt = 0; nt < 4; ++nt) {
      v16h bv = load_frag(vp + (size_t)(nt * 16) * T_DIM + s0, T_DIM, lane);
      o[nt] = wmma_f16f32(pa, bv, o[nt]);
    }
  }

  // ---- epilogue: normalize and store heads-concatenated [B,T,H*D] f16 ----
  const size_t obase = ((size_t)b * T_DIM + t0) * HD_DIM + (size_t)h * D_DIM;
#pragma unroll
  for (int nt = 0; nt < 4; ++nt)
#pragma unroll
    for (int r = 0; r < 8; ++r) {
      const int row = rbase + r;
      attn[obase + (size_t)row * HD_DIM + nt * 16 + col] =
          (_Float16)(o[nt][r] / lrow[r]);
    }
}

// ---------------- output projection: [B*T,1024] x [1024,1024] -> f32 ----------------
__global__ void __launch_bounds__(256) out_proj_kernel(
    const _Float16* __restrict__ attn,  // [B*T, HD]
    const _Float16* __restrict__ Wot,   // [E, HD] (Wo transposed)
    float* __restrict__ out) {          // [B*T, E]
  const int lane = threadIdx.x & 31;
  const int wave = blockIdx.x * 8 + (threadIdx.x >> 5);
  const int ct = wave & 15;   // 16 col-tiles of 64
  const int rt = wave >> 4;   // 512 row-tiles of 16
  if (rt >= (B_DIM * T_DIM) / 16) return;

  const _Float16* arow = attn + (size_t)rt * 16 * HD_DIM;
  const _Float16* wbase = Wot + (size_t)(ct * 64) * HD_DIM;

  v8f acc[4];
#pragma unroll
  for (int nt = 0; nt < 4; ++nt) acc[nt] = vzero8f();

  for (int kk = 0; kk < HD_DIM; kk += 32) {
    v16h a = load_frag(arow + kk, HD_DIM, lane);
#pragma unroll
    for (int nt = 0; nt < 4; ++nt) {
      v16h bw = load_frag(wbase + (size_t)(nt * 16) * HD_DIM + kk, HD_DIM, lane);
      acc[nt] = wmma_f16f32(a, bw, acc[nt]);
    }
  }

  const int col = lane & 15;
  const int rbase = (lane >> 4) * 8;
#pragma unroll
  for (int nt = 0; nt < 4; ++nt)
#pragma unroll
    for (int r = 0; r < 8; ++r)
      out[((size_t)rt * 16 + rbase + r) * E_DIM + ct * 64 + nt * 16 + col] =
          acc[nt][r];
}

// ---------------- host launcher ----------------
extern "C" void kernel_launch(void* const* d_in, const int* in_sizes, int n_in,
                              void* d_out, int out_size, void* d_ws, size_t ws_size,
                              hipStream_t stream) {
  (void)in_sizes; (void)n_in; (void)out_size; (void)ws_size;
  const float* x  = (const float*)d_in[0];
  const float* Wq = (const float*)d_in[1];
  const float* Wk = (const float*)d_in[2];
  const float* Wv = (const float*)d_in[3];
  const float* Wo = (const float*)d_in[4];
  float* out = (float*)d_out;

  char* ws = (char*)d_ws;
  size_t off = 0;
  auto take = [&](size_t bytes) -> _Float16* {
    _Float16* p = (_Float16*)(ws + off);
    off += (bytes + 255) & ~(size_t)255;
    return p;
  };
  _Float16* xh   = take((size_t)B_DIM * T_DIM * E_DIM * 2);          // 16 MB
  _Float16* Wqt  = take((size_t)H_DIM * D_DIM * E_DIM * 2);          // 2 MB
  _Float16* Wkt  = take((size_t)H_DIM * D_DIM * E_DIM * 2);
  _Float16* Wvt  = take((size_t)H_DIM * D_DIM * E_DIM * 2);
  _Float16* Wot  = take((size_t)E_DIM * HD_DIM * 2);                 // 2 MB
  _Float16* qbuf = take((size_t)B_DIM * H_DIM * T_DIM * D_DIM * 2);  // 16 MB
  _Float16* kbuf = take((size_t)B_DIM * H_DIM * T_DIM * D_DIM * 2);
  _Float16* vtb  = take((size_t)B_DIM * H_DIM * T_DIM * D_DIM * 2);
  _Float16* attn = take((size_t)B_DIM * T_DIM * HD_DIM * 2);         // 16 MB

  {
    int n = B_DIM * T_DIM * E_DIM;  // 8 Mi elems
    cvt_f32_to_f16<<<n / 256, 256, 0, stream>>>(x, xh, n);
  }
  {
    int blocks = (H_DIM * D_DIM * E_DIM) / 256;  // 4096
    cvt_tr_hed<<<blocks, 256, 0, stream>>>(Wq, Wqt);
    cvt_tr_hed<<<blocks, 256, 0, stream>>>(Wk, Wkt);
    cvt_tr_hed<<<blocks, 256, 0, stream>>>(Wv, Wvt);
    cvt_tr_wo<<<(E_DIM * HD_DIM) / 256, 256, 0, stream>>>(Wo, Wot);
  }
  {
    int waves = B_DIM * H_DIM * (T_DIM / 16);  // 8192 waves, 8 per block
    qkv_kernel<<<waves / 8, 256, 0, stream>>>(xh, Wqt, Wkt, Wvt, qbuf, kbuf, vtb);
    flash_attn_kernel<<<waves / 8, 256, 0, stream>>>(qbuf, kbuf, vtb, attn);
  }
  {
    int waves = ((B_DIM * T_DIM) / 16) * (E_DIM / 64);  // 8192
    out_proj_kernel<<<waves / 8, 256, 0, stream>>>(attn, Wot, out);
  }
}
